// DecoderModel_30262339567808
// MI455X (gfx1250) — matrix operand order repaired and checked by reading, not compiled, not run on hardware
//
#include <hip/hip_runtime.h>
#include <math.h>

typedef __attribute__((ext_vector_type(16))) _Float16 v16h;
typedef __attribute__((ext_vector_type(8)))  _Float16 v8h;
typedef __attribute__((ext_vector_type(4)))  _Float16 v4h;
typedef __attribute__((ext_vector_type(8)))  float    v8f;
typedef unsigned int u32x4 __attribute__((ext_vector_type(4)));
typedef int          i32x8 __attribute__((ext_vector_type(8)));
typedef int          i32x4 __attribute__((ext_vector_type(4)));

#define S_DIM 4096
#define FEA   256
#define H_DIM 1024
#define O_DIM 4096
#define KCH   256          // K-halves staged per TDM chunk (64 rows x 256 x 2B = 32 KB)

// ---------------- f32 -> f16 pre-conversion (one pass over each weight/input) --
__global__ __launch_bounds__(256) void cvt_f32_f16_kernel(const float* __restrict__ src,
                                                          _Float16* __restrict__ dst,
                                                          int n4) {
  const int i = blockIdx.x * 256 + threadIdx.x;
  if (i < n4) {
    const float4 v = ((const float4*)src)[i];
    v4h o;
    o[0] = (_Float16)v.x; o[1] = (_Float16)v.y;
    o[2] = (_Float16)v.z; o[3] = (_Float16)v.w;
    ((v4h*)dst)[i] = o;
  }
}

// ---------------- WMMA fragment loaders (wave32, 16x16x32 f16) ----------------
// A (16x32): lane&15 = M; K = k0 + 16*(lane>>4) + e (matches B)
__device__ inline v16h load_a_f16(const _Float16* __restrict__ A, int lda, int row,
                                  int k0, int lane) {
  const int kb = k0 + ((lane >> 4) << 4);
  const _Float16* p = A + (size_t)row * lda + kb;
  v8h lo = *(const v8h*)(p);
  v8h hi = *(const v8h*)(p + 8);
  v16h a;
#pragma unroll
  for (int e = 0; e < 8; ++e) { a[e] = lo[e]; a[8 + e] = hi[e]; }
  return a;
}

// B (32x16): lane&15 = N; K = k0 + 16*(lane>>4) + e -> one contiguous 32B load
__device__ inline v16h load_b_f16(const _Float16* __restrict__ B, int ldb, int nrow,
                                  int k0, int lane) {
  const int kb = k0 + ((lane >> 4) << 4);
  return *(const v16h*)(B + (size_t)nrow * ldb + kb);
}

__device__ inline float sigf(float x) { return 1.0f / (1.0f + __expf(-x)); }

// ---------------- Kernel 1: gates GEMM (f-gate skipped) + fused activation ----
// h = sig(o)*tanh( sig(i)*tanh(g) ), gates = inp @ W_ih.T + (b_ih + b_hh)
__global__ __launch_bounds__(128) void gates_h_kernel(
    const _Float16* __restrict__ inp16, const _Float16* __restrict__ wih16,
    const float* __restrict__ b_ih, const float* __restrict__ b_hh,
    _Float16* __restrict__ h16) {
  const int lane = threadIdx.x & 31;
  const int wave = threadIdx.x >> 5;
  const int l15  = lane & 15;
  const int rt = blockIdx.x;               // row tile 0..255
  const int ct = blockIdx.y * 4 + wave;    // h-col tile 0..63

  const int arow = rt * 16 + l15;
  const int ncol = ct * 16 + l15;
  v8f acc_i = {}; v8f acc_g = {}; v8f acc_o = {};

#pragma unroll
  for (int k0 = 0; k0 < FEA; k0 += 32) {
    v16h a  = load_a_f16(inp16, FEA, arow, k0, lane);
    v16h bi = load_b_f16(wih16, FEA, 0 * H_DIM + ncol, k0, lane);
    v16h bg = load_b_f16(wih16, FEA, 2 * H_DIM + ncol, k0, lane);
    v16h bo = load_b_f16(wih16, FEA, 3 * H_DIM + ncol, k0, lane);
    acc_i = __builtin_amdgcn_wmma_f32_16x16x32_f16(false, a, false, bi, (short)0, acc_i, false, false);
    acc_g = __builtin_amdgcn_wmma_f32_16x16x32_f16(false, a, false, bg, (short)0, acc_g, false, false);
    acc_o = __builtin_amdgcn_wmma_f32_16x16x32_f16(false, a, false, bo, (short)0, acc_o, false, false);
  }

  const int col = ct * 16 + l15;
  const float bi_s = b_ih[col]             + b_hh[col];
  const float bg_s = b_ih[2 * H_DIM + col] + b_hh[2 * H_DIM + col];
  const float bo_s = b_ih[3 * H_DIM + col] + b_hh[3 * H_DIM + col];
  const int mbase = rt * 16 + ((lane >> 4) << 3);
#pragma unroll
  for (int r = 0; r < 8; ++r) {
    float gi = acc_i[r] + bi_s;
    float gg = acc_g[r] + bg_s;
    float go = acc_o[r] + bo_s;
    float c  = sigf(gi) * tanhf(gg);
    float hv = sigf(go) * tanhf(c);
    h16[(size_t)(mbase + r) * H_DIM + col] = (_Float16)hv;
  }
}

// ---------------- Kernel 2: logits = h @ W2.T + b2, B staged via TDM ----------
// Block = 4 waves on 4 consecutive row-tiles sharing one 64-column W2 panel.
// TDM loads 64 rows x KCH halves of W2 (32 KB) into LDS per chunk.
__global__ __launch_bounds__(128) void logits_kernel(
    const _Float16* __restrict__ h16, const _Float16* __restrict__ w2h,
    const float* __restrict__ b2, float* __restrict__ logits) {
  __shared__ _Float16 bpanel[64 * KCH];    // 32 KB; only shared object -> LDS offset 0

  const int lane = threadIdx.x & 31;
  const int wave = threadIdx.x >> 5;
  const int l15  = lane & 15;
  const int rt    = blockIdx.x * 4 + wave; // row tile 0..255
  const int nbase = blockIdx.y * 64;       // output-column strip

  const int arow = rt * 16 + l15;
  v8f acc0 = {}, acc1 = {}, acc2 = {}, acc3 = {};

  for (int kc = 0; kc < H_DIM; kc += KCH) {
    __syncthreads();                       // previous chunk fully consumed
    if (threadIdx.x < 32) {
      // ---- Tensor DMA descriptor (2D tile: KCH x 64 rows of f16) ----
      const unsigned long long ga =
          (unsigned long long)(const void*)(w2h + (size_t)nbase * H_DIM + kc);
      u32x4 g0; i32x8 g1;
      i32x4 g2 = {0, 0, 0, 0};
      i32x4 g3 = {0, 0, 0, 0};
      i32x8 g4 = {0, 0, 0, 0, 0, 0, 0, 0};
      g0[0] = 1u;                                          // count=1, user mode
      g0[1] = 0u;                                          // lds_addr = bpanel (offset 0)
      g0[2] = (unsigned)(ga & 0xffffffffull);              // global_addr[31:0]
      g0[3] = (unsigned)((ga >> 32) & 0x01ffffffull) | (2u << 30); // addr[56:32] | type=2
      g1[0] = 0x10000;                                     // data_size=2B; no multicast
      g1[1] = (KCH & 0xffff) << 16;                        // tensor_dim0 = KCH
      g1[2] = (64 & 0xffff) << 16;                         // tensor_dim1 = 64
      g1[3] = (KCH & 0xffff) << 16;                        // tile_dim0 = KCH
      g1[4] = 64;                                          // tile_dim1 = 64
      g1[5] = H_DIM;                                       // tensor_dim0_stride = 1024
      g1[6] = 0; g1[7] = 0;
      __builtin_amdgcn_tensor_load_to_lds(g0, g1, g2, g3, g4, 0);
      __builtin_amdgcn_s_wait_tensorcnt(0);
    }
    __syncthreads();                       // panel visible to all 4 waves

#pragma unroll
    for (int k0 = 0; k0 < KCH; k0 += 32) {
      v16h a = load_a_f16(h16, H_DIM, arow, kc + k0, lane);
      const int kfrag = k0 + ((lane >> 4) << 4);
      v16h b0 = *(const v16h*)(&bpanel[( 0 + l15) * KCH + kfrag]);
      v16h b1 = *(const v16h*)(&bpanel[(16 + l15) * KCH + kfrag]);
      v16h b2v= *(const v16h*)(&bpanel[(32 + l15) * KCH + kfrag]);
      v16h b3 = *(const v16h*)(&bpanel[(48 + l15) * KCH + kfrag]);
      acc0 = __builtin_amdgcn_wmma_f32_16x16x32_f16(false, a, false, b0, (short)0, acc0, false, false);
      acc1 = __builtin_amdgcn_wmma_f32_16x16x32_f16(false, a, false, b1, (short)0, acc1, false, false);
      acc2 = __builtin_amdgcn_wmma_f32_16x16x32_f16(false, a, false, b2v,(short)0, acc2, false, false);
      acc3 = __builtin_amdgcn_wmma_f32_16x16x32_f16(false, a, false, b3, (short)0, acc3, false, false);
    }
  }

  const int mbase = rt * 16 + ((lane >> 4) << 3);
  v8f* accs[4] = {&acc0, &acc1, &acc2, &acc3};
#pragma unroll
  for (int j = 0; j < 4; ++j) {
    const int col = nbase + j * 16 + l15;
    const float bias = b2[col];
#pragma unroll
    for (int r = 0; r < 8; ++r)
      logits[(size_t)(mbase + r) * O_DIM + col] = (*accs[j])[r] + bias;
  }
}

// ---------------- Kernel 3: per-row logsumexp (replaces softmax) --------------
__global__ __launch_bounds__(256) void lse_kernel(const float* __restrict__ logits,
                                                  float* __restrict__ lse) {
  __shared__ float red[256];
  const int row = blockIdx.x;
  const int tid = threadIdx.x;
  const float* p = logits + (size_t)row * O_DIM;

  float m = -INFINITY;
  for (int c = tid; c < O_DIM; c += 256) m = fmaxf(m, p[c]);
  red[tid] = m; __syncthreads();
  for (int s = 128; s > 0; s >>= 1) {
    if (tid < s) red[tid] = fmaxf(red[tid], red[tid + s]);
    __syncthreads();
  }
  m = red[0]; __syncthreads();

  float sum = 0.f;
  for (int c = tid; c < O_DIM; c += 256) sum += __expf(p[c] - m);
  red[tid] = sum; __syncthreads();
  for (int s = 128; s > 0; s >>= 1) {
    if (tid < s) red[tid] += red[tid + s];
    __syncthreads();
  }
  if (tid == 0) lse[row] = m + __logf(red[0]);
}

// ---------------- Kernel 4: sequential greedy scan ----------------------------
// argmax(probs - mask) == argmax(logits over unchosen); logp = max_logit - lse.
__global__ __launch_bounds__(1024) void scan_kernel(const float* __restrict__ logits,
                                                    const float* __restrict__ lse,
                                                    int* __restrict__ out_idx,
                                                    float* __restrict__ out_logp) {
  __shared__ unsigned int maskw[O_DIM / 32];
  __shared__ float wv[32];
  __shared__ int   wi[32];
  const int tid  = threadIdx.x;
  const int lane = tid & 31;
  const int wave = tid >> 5;

  for (int i = tid; i < O_DIM / 32; i += 1024) maskw[i] = 0u;
  __syncthreads();

  for (int t = 0; t < S_DIM; ++t) {
    const float* row = logits + (size_t)t * O_DIM;
    float best = -INFINITY; int bidx = 0;
#pragma unroll
    for (int q = 0; q < 4; ++q) {
      const int c = tid + q * 1024;
      const float v = row[c];
      const bool taken = (maskw[c >> 5] >> (c & 31)) & 1u;
      if (!taken && v > best) { best = v; bidx = c; }
    }
#pragma unroll
    for (int off = 16; off > 0; off >>= 1) {
      float ov = __shfl_down(best, off, 32);
      int   oi = __shfl_down(bidx, off, 32);
      if (ov > best) { best = ov; bidx = oi; }
    }
    if (lane == 0) { wv[wave] = best; wi[wave] = bidx; }
    __syncthreads();
    if (wave == 0) {
      float v2 = wv[lane]; int i2 = wi[lane];
#pragma unroll
      for (int off = 16; off > 0; off >>= 1) {
        float ov = __shfl_down(v2, off, 32);
        int   oi = __shfl_down(i2, off, 32);
        if (ov > v2) { v2 = ov; i2 = oi; }
      }
      if (lane == 0) {
        out_idx[t]  = i2;
        out_logp[t] = v2 - lse[t];
        maskw[i2 >> 5] |= (1u << (i2 & 31));
      }
    }
    __syncthreads();
  }
}

// ---------------- host-side launcher ------------------------------------------
extern "C" void kernel_launch(void* const* d_in, const int* in_sizes, int n_in,
                              void* d_out, int out_size, void* d_ws, size_t ws_size,
                              hipStream_t stream) {
  const float* inp  = (const float*)d_in[0];
  const float* W_ih = (const float*)d_in[1];
  const float* b_ih = (const float*)d_in[2];
  const float* W_hh = (const float*)d_in[3];  (void)W_hh;  // only b_hh affects math
  const float* b_hh = (const float*)d_in[4];
  const float* W2   = (const float*)d_in[5];
  const float* b2   = (const float*)d_in[6];

  char* ws = (char*)d_ws;
  size_t off = 0;
  _Float16* inp16 = (_Float16*)(ws + off); off += (size_t)S_DIM * FEA   * 2;   // 2 MB
  _Float16* wih16 = (_Float16*)(ws + off); off += (size_t)4 * H_DIM * FEA * 2; // 2 MB
  _Float16* w2h   = (_Float16*)(ws + off); off += (size_t)O_DIM * H_DIM * 2;   // 8 MB
  _Float16* h16   = (_Float16*)(ws + off); off += (size_t)S_DIM * H_DIM * 2;   // 8 MB
  float*    logits= (float*)   (ws + off); off += (size_t)S_DIM * O_DIM * 4;   // 64 MB
  float*    lse   = (float*)   (ws + off);                                     // 16 KB

  int*   out_idx  = (int*)d_out;            // tour_idx  (4096 int32)
  float* out_logp = (float*)d_out + S_DIM;  // tour_logp (4096 f32)

  { const int n4 = S_DIM * FEA / 4;
    cvt_f32_f16_kernel<<<(n4 + 255) / 256, 256, 0, stream>>>(inp, inp16, n4); }
  { const int n4 = 4 * H_DIM * FEA / 4;
    cvt_f32_f16_kernel<<<(n4 + 255) / 256, 256, 0, stream>>>(W_ih, wih16, n4); }
  { const int n4 = O_DIM * H_DIM / 4;
    cvt_f32_f16_kernel<<<(n4 + 255) / 256, 256, 0, stream>>>(W2, w2h, n4); }

  gates_h_kernel<<<dim3(S_DIM / 16, 16), 128, 0, stream>>>(inp16, wih16, b_ih, b_hh, h16);
  logits_kernel <<<dim3(S_DIM / 64, O_DIM / 64), 128, 0, stream>>>(h16, w2h, b2, logits);
  lse_kernel    <<<dim3(S_DIM), 256, 0, stream>>>(logits, lse);
  scan_kernel   <<<1, 1024, 0, stream>>>(logits, lse, out_idx, out_logp);
}
